// SirenMLP_67963562492064
// MI455X (gfx1250) — matrix operand order repaired and checked by reading, not compiled, not run on hardware
//
#include <hip/hip_runtime.h>
#include <hip/hip_fp16.h>

typedef __attribute__((ext_vector_type(16))) _Float16 v16h;
typedef __attribute__((ext_vector_type(8)))  _Float16 v8h;
typedef __attribute__((ext_vector_type(4)))  _Float16 v4h;
typedef __attribute__((ext_vector_type(8)))  float    v8f;
typedef __attribute__((ext_vector_type(4)))  float    v4f;
typedef __attribute__((ext_vector_type(4)))  unsigned int u32x4;
typedef __attribute__((ext_vector_type(8)))  int      i32x8;
typedef __attribute__((ext_vector_type(4)))  int      i32x4;

#define HID            256
#define ROWS_PER_BLOCK 128          // 8 waves x 16 rows
#define N_HIDDEN_L     4
#define OMEGA          30.0f
#define CHUNK_ROWS     64           // W rows per TDM chunk
#define CHUNK_HALVES   (CHUNK_ROWS * HID)   // 16384 halves = 32 KB
#define CHUNK_ELEMS8   (CHUNK_HALVES / 4)   // 4096 8-byte elements per chunk

#if __has_builtin(__builtin_amdgcn_tensor_load_to_lds) && \
    __has_builtin(__builtin_amdgcn_s_wait_tensorcnt)
#define HAVE_TDM 1
#else
#define HAVE_TDM 0
#endif

union V16 { v16h v; v8h h[2]; };

// ---------- one-shot weight conversion fp32 -> f16 into workspace ----------
__global__ __launch_bounds__(256) void siren_convert_w_kernel(
    const float* __restrict__ w, _Float16* __restrict__ o, int n4)
{
    const int i = blockIdx.x * 256 + threadIdx.x;
    if (i < n4) {
        v4f f = *(const v4f*)(w + (size_t)i * 4);
        v4h h;
        #pragma unroll
        for (int e = 0; e < 4; ++e) h[e] = (_Float16)f[e];
        *(v4h*)(o + (size_t)i * 4) = h;
    }
}

#if HAVE_TDM
// Issue one TDM copy of a 32 KB contiguous f16 chunk (global -> LDS).
// D# per ISA 8.3/8.4: group0 = {count=1, lds_addr, global_addr, type=2};
// group1 = {data_size=8B, tensor_dim0=tile_dim0=4096 elems, 1 row, stride=4096}.
__device__ __forceinline__ void tdm_load_chunk(const _Float16* gsrc, unsigned lds_off)
{
    const unsigned long long ga = (unsigned long long)(uintptr_t)gsrc;
    u32x4 g0;
    g0[0] = 1u;                                             // count=1, user mode
    g0[1] = lds_off;                                        // lds_addr (bytes)
    g0[2] = (unsigned)ga;                                   // global_addr[31:0]
    g0[3] = ((unsigned)(ga >> 32) & 0x01FFFFFFu) | (2u << 30); // addr[56:32] | type=2
    i32x8 g1;
    g1[0] = (int)(3u << 16);                                // data_size = 8 bytes
    g1[1] = (int)((CHUNK_ELEMS8 & 0xFFFFu) << 16);          // tensor_dim0 lo16
    g1[2] = (int)(1u << 16);                                // tensor_dim0 hi=0 | tensor_dim1=1
    g1[3] = (int)((CHUNK_ELEMS8 & 0xFFFFu) << 16);          // tensor_dim1 hi=0 | tile_dim0
    g1[4] = 1;                                              // tile_dim1=1, tile_dim2=0
    g1[5] = CHUNK_ELEMS8;                                   // tensor_dim0_stride lo32
    g1[6] = 0;
    g1[7] = 0;
    const i32x4 z4 = {0, 0, 0, 0};
#if __clang_major__ >= 23
    i32x8 z8;
    #pragma unroll
    for (int e = 0; e < 8; ++e) z8[e] = 0;
    __builtin_amdgcn_tensor_load_to_lds(g0, g1, z4, z4, z8, 0);
#else
    __builtin_amdgcn_tensor_load_to_lds(g0, g1, z4, z4, 0);
#endif
}
#endif

// ---------- fully fused SIREN forward ----------
__global__ __launch_bounds__(256) void siren_fused_kernel(
    const float*    __restrict__ x,        // [N,3]
    const float*    __restrict__ w_first,  // [256,3]
    const float*    __restrict__ b_first,  // [256]
    const _Float16* __restrict__ wh16,     // [4,256,256] f16 (pre-converted)
    const float*    __restrict__ b_hidden, // [4,256]
    const float*    __restrict__ w_final,  // [3,256]
    const float*    __restrict__ b_final,  // [3]
    float*          __restrict__ out)      // [N,3]
{
    __shared__ _Float16 hbuf[ROWS_PER_BLOCK * HID]; // 64 KB activation slab
    __shared__ _Float16 wbuf0[CHUNK_HALVES];        // 32 KB weight chunk (ping)
    __shared__ _Float16 wbuf1[CHUNK_HALVES];        // 32 KB weight chunk (pong)
    __shared__ float    xbuf[ROWS_PER_BLOCK * 3];

    const int tid  = threadIdx.x;
    const int lane = tid & 31;
    const int wave = tid >> 5;
    const int row0 = blockIdx.x * ROWS_PER_BLOCK;

    // stage x
    for (int i = tid; i < ROWS_PER_BLOCK * 3; i += 256)
        xbuf[i] = x[row0 * 3 + i];

#if HAVE_TDM
    // prologue: DMA the first two weight chunks while the first layer computes
    const unsigned wb0_off = (unsigned)(uintptr_t)&wbuf0[0];
    const unsigned wb1_off = (unsigned)(uintptr_t)&wbuf1[0];
    if (wave == 0) {
        tdm_load_chunk(wh16,                wb0_off);
        tdm_load_chunk(wh16 + CHUNK_HALVES, wb1_off);
    }
#endif
    __syncthreads();

    // ---- first sine layer (K=3, scalar): thread owns column c = tid ----
    {
        const int   c  = tid;
        const float w0 = w_first[c * 3 + 0];
        const float w1 = w_first[c * 3 + 1];
        const float w2 = w_first[c * 3 + 2];
        const float bb = b_first[c];
        #pragma unroll 4
        for (int r = 0; r < ROWS_PER_BLOCK; ++r) {
            float t = fmaf(xbuf[r * 3 + 2], w2,
                      fmaf(xbuf[r * 3 + 1], w1,
                      fmaf(xbuf[r * 3 + 0], w0, bb)));
            hbuf[r * HID + c] = (_Float16)__sinf(OMEGA * t);
        }
    }

    // ---- hidden sine layers via WMMA f32<-f16, TDM-pipelined weights ----
    const int m    = lane & 15;
    const int hi   = lane >> 4;
    const int arow = wave * 16 + m;

    for (int l = 0; l < N_HIDDEN_L; ++l) {
        const float* bl = b_hidden + l * HID;

        v8f acc[16];
        #pragma unroll
        for (int i = 0; i < 16; ++i)
            #pragma unroll
            for (int e = 0; e < 8; ++e) acc[i][e] = 0.0f;

        #pragma unroll
        for (int q = 0; q < 4; ++q) {                // chunk c = 4*l + q
            _Float16* wb = (q & 1) ? wbuf1 : wbuf0;

#if HAVE_TDM
            // chunk c must have landed; keep one DMA (chunk c+1) in flight
            if (wave == 0) {
                if (q == 3 && l == N_HIDDEN_L - 1)
                    __builtin_amdgcn_s_wait_tensorcnt(0);
                else
                    __builtin_amdgcn_s_wait_tensorcnt(1);
            }
#else
            // fallback: cooperative synchronous copy of chunk c
            {
                const _Float16* gs = wh16 + (size_t)(4 * l + q) * CHUNK_HALVES;
                #pragma unroll
                for (int j = 0; j < 8; ++j) {
                    const int i = j * 2048 + tid * 8;
                    *(v8h*)(wb + i) = *(const v8h*)(gs + i);
                }
            }
#endif
            __syncthreads();                          // chunk c visible to all waves

            #pragma unroll
            for (int kt = 0; kt < 8; ++kt) {
                const int k0 = kt * 32;
                // A (16x32 f16) per ISA 7.12.2 lane layout
                V16 A;
                const _Float16* ap = hbuf + arow * HID + k0 + hi * 8;
                A.h[0] = *(const v8h*)(ap);
                A.h[1] = *(const v8h*)(ap + 16);

                #pragma unroll
                for (int nt = 0; nt < 4; ++nt) {      // output cols [64q, 64q+64)
                    V16 B;
                    const _Float16* bp = wb + (nt * 16 + m) * HID + k0 + hi * 8;
                    B.h[0] = *(const v8h*)(bp);
                    B.h[1] = *(const v8h*)(bp + 16);

                    acc[q * 4 + nt] = __builtin_amdgcn_wmma_f32_16x16x32_f16(
                        false, A.v, false, B.v, (short)0, acc[q * 4 + nt],
                        false, false);
                }
            }
            __syncthreads();                          // all waves done reading wb

#if HAVE_TDM
            // refill the buffer just released with chunk c+2
            if (wave == 0) {
                const int c2 = 4 * l + q + 2;
                if (c2 < 4 * N_HIDDEN_L)
                    tdm_load_chunk(wh16 + (size_t)c2 * CHUNK_HALVES,
                                   (q & 1) ? wb1_off : wb0_off);
            }
#endif
        }

        // activation + writeback (rows private to this wave; no barrier needed)
        #pragma unroll
        for (int nt = 0; nt < 16; ++nt) {
            const int   n  = nt * 16 + m;
            const float bn = bl[n];
            _Float16* hp = hbuf + (wave * 16 + hi * 8) * HID + n;
            v8f a = acc[nt];
            #pragma unroll
            for (int v = 0; v < 8; ++v)
                hp[v * HID] = (_Float16)__sinf(OMEGA * (a[v] + bn));
        }
    }
    __syncthreads();

    // ---- final linear layer (N=3, scalar): thread r handles one point ----
    if (tid < ROWS_PER_BLOCK) {
        const int r = tid;
        float a0 = b_final[0], a1 = b_final[1], a2 = b_final[2];
        const _Float16* hp = hbuf + r * HID;
        #pragma unroll 8
        for (int k = 0; k < HID; ++k) {
            const float h = (float)hp[k];
            a0 = fmaf(h, w_final[k],           a0);
            a1 = fmaf(h, w_final[HID + k],     a1);
            a2 = fmaf(h, w_final[2 * HID + k], a2);
        }
        float* op = out + (size_t)(row0 + r) * 3;
        op[0] = a0; op[1] = a1; op[2] = a2;
    }
}

extern "C" void kernel_launch(void* const* d_in, const int* in_sizes, int n_in,
                              void* d_out, int out_size, void* d_ws, size_t ws_size,
                              hipStream_t stream) {
    (void)in_sizes; (void)n_in; (void)ws_size;
    const float* x        = (const float*)d_in[0];
    const float* w_first  = (const float*)d_in[1];
    const float* b_first  = (const float*)d_in[2];
    const float* w_hidden = (const float*)d_in[3];
    const float* b_hidden = (const float*)d_in[4];
    const float* w_final  = (const float*)d_in[5];
    const float* b_final  = (const float*)d_in[6];
    float* out = (float*)d_out;
    _Float16* wh16 = (_Float16*)d_ws;   // 4*256*256 f16 = 512 KB scratch

    // one-shot fp32 -> f16 weight conversion (65536 float4 groups)
    const int n4 = N_HIDDEN_L * HID * HID / 4;
    siren_convert_w_kernel<<<(n4 + 255) / 256, 256, 0, stream>>>(w_hidden, wh16, n4);

    const int n_pts  = out_size / 3;                 // 262144
    const int blocks = n_pts / ROWS_PER_BLOCK;       // 2048
    siren_fused_kernel<<<blocks, 256, 0, stream>>>(
        x, w_first, b_first, wh16, b_hidden, w_final, b_final, out);
}
